// Decoder_26233660244038
// MI455X (gfx1250) — matrix operand order repaired and checked by reading, not compiled, not run on hardware
//
#include <hip/hip_runtime.h>
#include <stdint.h>
#include <math.h>

// ---------------- dims ----------------
#define BATCH 32
#define AGT   21      // A
#define MMAP  400     // M
#define LTOT  421     // L = A + M
#define DMODEL 256
#define NHEAD 8
#define DHEAD 32
#define TSTEP 21
#define SDIM  8
#define RREF  6
#define PPTS  50
#define FHOR  80
#define KMODE 6
#define ENVL  442     // A + L
#define NEGV  -1000000000.0f

typedef __attribute__((ext_vector_type(8)))  float  v8f;
typedef __attribute__((ext_vector_type(16))) __bf16 v16bf;

__device__ __forceinline__ unsigned short f2bf(float f) {
  unsigned x = __float_as_uint(f);
  return (unsigned short)((x + 0x7fffu + ((x >> 16) & 1u)) >> 16);
}

// ---------------- generic batched WMMA GEMM ----------------
// C[z] = act( alpha * A[z] @ B[z] + bias + add[row/addDiv] )
// z decomposed as (zb, zh) with zb = z / Hdim, zh = z % Hdim.
struct GemmParams {
  const float* A;  int64_t lda, sAb, sAh;
  const float* Bm; int64_t ldb, sBb, sBh; int transB;
  float*       C;  int64_t ldc, sCb, sCh;
  const float* bias;
  const float* add; int64_t addld; int addDiv;
  int act;            // 0 none, 1 relu, 2 elu
  float alpha;
  int M, N, K, Hdim;
};

// LDS layouts (fragment-order so lanes read contiguous b128s):
//  As[r][k] : k contiguous -> lane<16 reads k 0..15 (bytes [0,32)),
//             lane>=16 reads k 16..31 (bytes [32,64)) of each 64B row.
//  Bs[n][k] : transposed; lane<16 reads k 0..15, lane>=16 reads k 16..31.
__device__ __forceinline__ int permA(int c) {           // c in [0,32)
  int pg = c >> 3;                                      // identity: g -> g
  return (pg << 3) | (c & 7);
}

__global__ __launch_bounds__(256) void wmma_gemm(GemmParams p) {
  const int zb = blockIdx.z / p.Hdim, zh = blockIdx.z % p.Hdim;
  const float* A  = p.A  + zb * p.sAb + zh * p.sAh;
  const float* Bm = p.Bm + zb * p.sBb + zh * p.sBh;
  float*       C  = p.C  + zb * p.sCb + zh * p.sCh;
  const int tileM = blockIdx.y * 64, tileN = blockIdx.x * 64;
  const int tid  = threadIdx.x;
  const int lane = tid & 31, wv = tid >> 5;
  const int wr = wv >> 1, wc = wv & 1;            // 4x2 wave grid; each wave: 16x32
  __shared__ __attribute__((aligned(16))) unsigned short As[64 * 32];
  __shared__ __attribute__((aligned(16))) unsigned short Bs[64 * 32];
  v8f acc0 = {0,0,0,0,0,0,0,0};
  v8f acc1 = {0,0,0,0,0,0,0,0};

  const bool ldaAligned = ((p.lda & 3) == 0);
  const bool ldbAligned = ((p.ldb & 3) == 0);

  for (int kk = 0; kk < p.K; kk += 32) {
    const bool fullK = (kk + 32 <= p.K);
    // ---- stage A tile 64x32 (fp32 -> bf16, fragment-permuted) ----
    if (fullK && ldaAligned) {
      int r = tid >> 2, j = tid & 3;                 // j: 8-wide k group
      int gm = tileM + r;
      union { unsigned short us[8]; uint4 q; } t;
      if (gm < p.M) {
        const float* src = A + (int64_t)gm * p.lda + kk + j * 8;
        float4 x0 = ((const float4*)src)[0];
        float4 x1 = ((const float4*)src)[1];
        t.us[0] = f2bf(x0.x); t.us[1] = f2bf(x0.y);
        t.us[2] = f2bf(x0.z); t.us[3] = f2bf(x0.w);
        t.us[4] = f2bf(x1.x); t.us[5] = f2bf(x1.y);
        t.us[6] = f2bf(x1.z); t.us[7] = f2bf(x1.w);
      } else {
        t.q = make_uint4(0, 0, 0, 0);
      }
      int pg = j;
      *(uint4*)(As + r * 32 + pg * 8) = t.q;
    } else {
#pragma unroll
      for (int i = 0; i < 8; i++) {
        int e = tid + i * 256;
        int r = e >> 5, c = e & 31;
        int gm = tileM + r, gk = kk + c;
        float v = (gm < p.M && gk < p.K) ? A[(int64_t)gm * p.lda + gk] : 0.f;
        As[r * 32 + permA(c)] = f2bf(v);
      }
    }
    // ---- stage B tile 32x64 into Bs[n][k] ----
    if (fullK && ldbAligned && (tileN + 64 <= p.N)) {
      if (p.transB) {
        int n = tid >> 2, j = tid & 3;
        const float* src = Bm + (int64_t)(tileN + n) * p.ldb + kk + j * 8;
        float4 x0 = ((const float4*)src)[0];
        float4 x1 = ((const float4*)src)[1];
        union { unsigned short us[8]; uint4 q; } t;
        t.us[0] = f2bf(x0.x); t.us[1] = f2bf(x0.y);
        t.us[2] = f2bf(x0.z); t.us[3] = f2bf(x0.w);
        t.us[4] = f2bf(x1.x); t.us[5] = f2bf(x1.y);
        t.us[6] = f2bf(x1.z); t.us[7] = f2bf(x1.w);
        *(uint4*)(Bs + n * 32 + j * 8) = t.q;
      } else {
        int k = tid >> 3, n0 = (tid & 7) * 8;
        const float* src = Bm + (int64_t)(kk + k) * p.ldb + tileN + n0;
        float4 x0 = ((const float4*)src)[0];
        float4 x1 = ((const float4*)src)[1];
        unsigned short us[8];
        us[0] = f2bf(x0.x); us[1] = f2bf(x0.y); us[2] = f2bf(x0.z); us[3] = f2bf(x0.w);
        us[4] = f2bf(x1.x); us[5] = f2bf(x1.y); us[6] = f2bf(x1.z); us[7] = f2bf(x1.w);
#pragma unroll
        for (int i = 0; i < 8; i++) Bs[(n0 + i) * 32 + k] = us[i];
      }
    } else {
#pragma unroll
      for (int i = 0; i < 8; i++) {
        int e = tid + i * 256;
        int k = e >> 6, n = e & 63;
        int gk = kk + k, gn = tileN + n;
        float v = 0.f;
        if (gk < p.K && gn < p.N)
          v = p.transB ? Bm[(int64_t)gn * p.ldb + gk] : Bm[(int64_t)gk * p.ldb + gn];
        Bs[n * 32 + k] = f2bf(v);
      }
    }
    __syncthreads();

    // ---- fragment loads: contiguous b128 pairs ----
    union FR { v16bf v; uint4 q[2]; } fa, fb0, fb1;
    const int rA = wr * 16 + (lane & 15);
    const int half = (lane < 16) ? 0 : 16;     // A: permuted half; B: k half
    const uint4* ap = (const uint4*)(As + rA * 32 + half);
    fa.q[0] = ap[0]; fa.q[1] = ap[1];
    const int nB = wc * 32 + (lane & 15);
    const uint4* bp0 = (const uint4*)(Bs + nB * 32 + half);
    fb0.q[0] = bp0[0]; fb0.q[1] = bp0[1];
    const uint4* bp1 = (const uint4*)(Bs + (nB + 16) * 32 + half);
    fb1.q[0] = bp1[0]; fb1.q[1] = bp1[1];

    acc0 = __builtin_amdgcn_wmma_f32_16x16x32_bf16(false, fa.v, false, fb0.v,
                                                   (short)0, acc0, false, false);
    acc1 = __builtin_amdgcn_wmma_f32_16x16x32_bf16(false, fa.v, false, fb1.v,
                                                   (short)0, acc1, false, false);
    __syncthreads();
  }

  const int adiv = (p.addDiv > 0) ? p.addDiv : 1;
#pragma unroll
  for (int e = 0; e < 8; e++) {
    int row = tileM + wr * 16 + ((lane < 16) ? e : 8 + e);
    if (row >= p.M) continue;
#pragma unroll
    for (int t = 0; t < 2; t++) {
      int col = tileN + wc * 32 + t * 16 + (lane & 15);
      if (col >= p.N) continue;
      float v = (t == 0 ? acc0[e] : acc1[e]) * p.alpha;
      if (p.bias) v += p.bias[col];
      if (p.add)  v += p.add[(int64_t)(row / adiv) * p.addld + col];
      if (p.act == 1)      v = fmaxf(v, 0.f);
      else if (p.act == 2) v = (v > 0.f) ? v : expm1f(v);
      C[(int64_t)row * p.ldc + col] = v;
    }
  }
}

// ---------------- fused mask + softmax over last dim ----------------
__global__ void softmax_mask(float* logits, int H, int Q, int Kv,
                             const unsigned char* mask) {
  int row = blockIdx.x;                 // (b*H + h)*Q + q
  int b = row / (H * Q);
  float* ptr = logits + (int64_t)row * Kv;
  __shared__ float red[256];
  int tid = threadIdx.x;
  float mx = -3.4e38f;
  for (int c = tid; c < Kv; c += 256) {
    float v = ptr[c];
    if (mask && mask[(int64_t)b * Kv + c]) v = NEGV;
    ptr[c] = v;
    mx = fmaxf(mx, v);
  }
  red[tid] = mx; __syncthreads();
  for (int s = 128; s > 0; s >>= 1) { if (tid < s) red[tid] = fmaxf(red[tid], red[tid + s]); __syncthreads(); }
  mx = red[0]; __syncthreads();
  float sum = 0.f;
  for (int c = tid; c < Kv; c += 256) { float e = expf(ptr[c] - mx); ptr[c] = e; sum += e; }
  red[tid] = sum; __syncthreads();
  for (int s = 128; s > 0; s >>= 1) { if (tid < s) red[tid] += red[tid + s]; __syncthreads(); }
  float inv = 1.f / red[0];
  for (int c = tid; c < Kv; c += 256) ptr[c] *= inv;
}

// ---------------- scalar GEMM (tiny K / tiny N) ----------------
__global__ void scalar_gemm(const float* X, int64_t lda, const float* W,
                            const float* bias, float* C, int64_t ldc,
                            int M, int N, int K, int act) {
  int64_t i = (int64_t)blockIdx.x * blockDim.x + threadIdx.x;
  if (i >= (int64_t)M * N) return;
  int row = (int)(i / N), col = (int)(i % N);
  float s = 0.f;
  for (int k = 0; k < K; k++) s += X[(int64_t)row * lda + k] * W[(int64_t)k * N + col];
  if (bias) s += bias[col];
  if (act == 1) s = fmaxf(s, 0.f);
  C[(int64_t)row * ldc + col] = s;
}

// ---------------- small utility kernels ----------------
__global__ void k_concat2(const float* a, const float* b, float* o, int rows, int d) {
  int64_t i = (int64_t)blockIdx.x * 256 + threadIdx.x;
  if (i >= (int64_t)rows * 2 * d) return;
  int row = (int)(i / (2 * d)), c = (int)(i % (2 * d));
  o[i] = (c < d) ? a[(int64_t)row * d + c] : b[(int64_t)row * d + (c - d)];
}
__global__ void k_add_bcast(const float* x, const float* v, float* o, int rows, int d) {
  int64_t i = (int64_t)blockIdx.x * 256 + threadIdx.x;
  if (i >= (int64_t)rows * d) return;
  o[i] = x[i] + v[i % d];
}
__global__ void k_traj_pack(const float* pred, float* traj) {
  int64_t i = (int64_t)blockIdx.x * 256 + threadIdx.x;
  if (i >= (int64_t)BATCH * AGT * KMODE * FHOR * 2) return;
  int64_t row = i / (FHOR * 2); int j = (int)(i % (FHOR * 2));
  traj[i] = pred[row * (FHOR * 4) + (j >> 1) * 4 + (j & 1)];
}
__global__ void k_maxpool(const float* in, float* out, int rows, int p, int d) {
  int64_t i = (int64_t)blockIdx.x * 256 + threadIdx.x;
  if (i >= (int64_t)rows * d) return;
  int row = (int)(i / d), c = (int)(i % d);
  float m = -3.4e38f;
  for (int j = 0; j < p; j++) m = fmaxf(m, in[((int64_t)row * p + j) * d + c]);
  out[i] = m;
}
__global__ void k_scores_softmax(const float* sc, float* sm, int rows) {
  int i = blockIdx.x * blockDim.x + threadIdx.x;
  if (i >= rows) return;
  float mx = -3.4e38f;
  for (int j = 0; j < KMODE; j++) mx = fmaxf(mx, sc[i * KMODE + j]);
  float e[KMODE], s = 0.f;
  for (int j = 0; j < KMODE; j++) { e[j] = expf(sc[i * KMODE + j] - mx); s += e[j]; }
  for (int j = 0; j < KMODE; j++) sm[i * KMODE + j] = e[j] / s;
}
__global__ void k_futures(const float* fut, const float* sm, float* o, int rowsBA, int d) {
  int64_t i = (int64_t)blockIdx.x * 256 + threadIdx.x;
  if (i >= (int64_t)rowsBA * d) return;
  int row = (int)(i / d), c = (int)(i % d);
  float s = 0.f;
  for (int j = 0; j < KMODE; j++)
    s += fut[((int64_t)row * KMODE + j) * d + c] * sm[row * KMODE + j];
  o[i] = s / (float)KMODE;
}
__global__ void k_env_pack(const float* fu, const float* enc, float* env, int d) {
  int64_t i = (int64_t)blockIdx.x * 256 + threadIdx.x;
  if (i >= (int64_t)BATCH * ENVL * d) return;
  int64_t row = i / d; int c = (int)(i % d);
  int b = (int)(row / ENVL), r = (int)(row % ENVL);
  env[i] = (r < AGT) ? fu[((int64_t)b * AGT + r) * d + c]
                     : enc[((int64_t)b * LTOT + (r - AGT)) * d + c];
}
__global__ void k_mask_pack(const unsigned char* am, const unsigned char* mm,
                            unsigned char* em) {
  int i = blockIdx.x * 256 + threadIdx.x;
  if (i >= BATCH * ENVL) return;
  int b = i / ENVL, r = i % ENVL;
  em[i] = (r < AGT) ? am[b * AGT + r] : mm[b * LTOT + (r - AGT)];
}
__global__ void k_refmask(const float* rp, float* rm) {
  int i = blockIdx.x * blockDim.x + threadIdx.x;
  if (i >= BATCH * RREF) return;
  const float* p = rp + (int64_t)i * (PPTS * 5);
  int z = 1;
  for (int j = 0; j < PPTS * 5; j++) if (p[j] != 0.f) { z = 0; break; }
  rm[i] = (float)z;
}
__global__ void k_score_r(const float* xr, const float* w, const float* b,
                          const float* rm, float* sr) {
  int i = blockIdx.x * blockDim.x + threadIdx.x;
  if (i >= BATCH * RREF) return;
  float s = b[0];
  const float* x = xr + (int64_t)i * DMODEL;
  for (int j = 0; j < DMODEL; j++) s += x[j] * w[j];
  sr[i] = (rm[i] != 0.f) ? NEGV : s;
}
__global__ void k_argmax_gather(const float* sr, const float* xr, float* ego) {
  int b = blockIdx.x;
  __shared__ int sidx;
  if (threadIdx.x == 0) {
    int bi = 0; float bv = sr[b * RREF];
    for (int j = 1; j < RREF; j++)
      if (sr[b * RREF + j] > bv) { bv = sr[b * RREF + j]; bi = j; }
    sidx = bi;
  }
  __syncthreads();
  for (int c = threadIdx.x; c < DMODEL; c += blockDim.x)
    ego[b * DMODEL + c] = xr[((int64_t)b * RREF + sidx) * DMODEL + c];
}
__global__ void k_dynamics(const float* ctrl, const float* actors, float* out) {
  int b = blockIdx.x * blockDim.x + threadIdx.x;
  if (b >= BATCH) return;
  const float* ic = actors + ((int64_t)b * AGT * TSTEP + (TSTEP - 1)) * SDIM; // actors[b,0,-1]
  float x0 = ic[0], y0 = ic[1], yaw0 = ic[2];
  float v0 = sqrtf(ic[3] * ic[3] + ic[4] * ic[4]);
  const float dt = 0.1f, TWO_PI = 6.28318530717958647692f;
  float sa = 0.f, syaw = 0.f, sx = 0.f, sy = 0.f;
  for (int f = 0; f < FHOR; f++) {
    float a = fminf(fmaxf(ctrl[(b * FHOR + f) * 2 + 0], -5.f), 5.f);
    float w = fminf(fmaxf(ctrl[(b * FHOR + f) * 2 + 1], -0.5f), 0.5f);
    sa += a * dt;
    float vel = fmaxf(v0 + sa, 0.f);
    syaw += w * vel * dt;
    float yaw = fmodf(yaw0 + syaw, TWO_PI);
    sx += vel * cosf(yaw) * dt;
    sy += vel * sinf(yaw) * dt;
    float* o = out + ((int64_t)b * FHOR + f) * 3;
    o[0] = x0 + sx; o[1] = y0 + sy; o[2] = yaw;
  }
}

// ---------------- host-side helpers ----------------
static void launch_gemm(hipStream_t s, const GemmParams& p, int z) {
  dim3 g((p.N + 63) / 64, (p.M + 63) / 64, z);
  wmma_gemm<<<g, 256, 0, s>>>(p);
}
static GemmParams GP() {
  GemmParams p{};                 // value-init: zero-fills all members
  p.alpha = 1.f; p.Hdim = 1; p.addDiv = 1;
  return p;
}

static void proj(hipStream_t st, const float* X, int rows, const float* W, float* Y,
                 const float* bias = nullptr, const float* add = nullptr,
                 int act = 0, int N = DMODEL, int Kd = DMODEL) {
  GemmParams p = GP();
  p.A = X; p.lda = Kd; p.Bm = W; p.ldb = N; p.C = Y; p.ldc = N;
  p.bias = bias; p.add = add; p.addld = N;
  p.act = act; p.M = rows; p.N = N; p.K = Kd;
  launch_gemm(st, p, 1);
}
// attention core: logits = alpha * Qh @ Kh^T ; softmax(mask) ; ctx = W @ Vh
static void attn_core(hipStream_t st, float* Q, float* Kb, float* V,
                      float* Log, float* Ctx, int Qlen, int Kv,
                      const unsigned char* mask) {
  GemmParams lp = GP();
  lp.A = Q; lp.lda = DMODEL; lp.sAb = (int64_t)Qlen * DMODEL; lp.sAh = DHEAD;
  lp.Bm = Kb; lp.ldb = DMODEL; lp.sBb = (int64_t)Kv * DMODEL; lp.sBh = DHEAD; lp.transB = 1;
  lp.C = Log; lp.ldc = Kv; lp.sCb = (int64_t)NHEAD * Qlen * Kv; lp.sCh = (int64_t)Qlen * Kv;
  lp.alpha = 0.17677669529663687f;   // 1/sqrt(32)
  lp.M = Qlen; lp.N = Kv; lp.K = DHEAD; lp.Hdim = NHEAD;
  launch_gemm(st, lp, BATCH * NHEAD);
  softmax_mask<<<BATCH * NHEAD * Qlen, 256, 0, st>>>(Log, NHEAD, Qlen, Kv, mask);
  GemmParams cp = GP();
  cp.A = Log; cp.lda = Kv; cp.sAb = (int64_t)NHEAD * Qlen * Kv; cp.sAh = (int64_t)Qlen * Kv;
  cp.Bm = V; cp.ldb = DMODEL; cp.sBb = (int64_t)Kv * DMODEL; cp.sBh = DHEAD;
  cp.C = Ctx; cp.ldc = DMODEL; cp.sCb = (int64_t)Qlen * DMODEL; cp.sCh = DHEAD;
  cp.M = Qlen; cp.N = DHEAD; cp.K = Kv; cp.Hdim = NHEAD;
  launch_gemm(st, cp, BATCH * NHEAD);
}
static inline int grid1d(int64_t n) { return (int)((n + 255) / 256); }

// ---------------- top level ----------------
extern "C" void kernel_launch(void* const* d_in, const int* in_sizes, int n_in,
                              void* d_out, int out_size, void* d_ws, size_t ws_size,
                              hipStream_t stream) {
  const float* actors      = (const float*)d_in[0];
  const float* encoding    = (const float*)d_in[1];
  const unsigned char* msk = (const unsigned char*)d_in[2];
  const unsigned char* mapm= (const unsigned char*)d_in[3];
  const unsigned char* actm= (const unsigned char*)d_in[4];
  const float* ref_paths   = (const float*)d_in[5];
  const float* P_[64];
  for (int i = 6; i < n_in && i < 64; i++) P_[i] = (const float*)d_in[i];
  const float *caWq=P_[6], *caWk=P_[7], *caWv=P_[8], *caWo=P_[9];
  const float *mmWq=P_[10],*mmWk=P_[11],*mmWv=P_[12],*mmWo=P_[13];
  const float *itWq=P_[14],*itWk=P_[15],*itWv=P_[16],*itWo=P_[17];
  const float *dlWq=P_[18],*dlWk=P_[19],*dlWv=P_[20],*dlWo=P_[21];
  const float *fuW1=P_[22],*fub1=P_[23],*fuW2=P_[24],*fub2=P_[25];
  const float *gWt=P_[26], *gbt=P_[27], *gWs=P_[28], *gbs=P_[29];
  const float *feWt=P_[30],*febt=P_[31],*feWx=P_[32],*febx=P_[33],*feWo=P_[34],*febo=P_[35];
  const float *rW1=P_[36], *rb1=P_[37], *rW2=P_[38], *rb2=P_[39];
  const float *dmW1=P_[40],*dmb1=P_[41],*dmW2=P_[42],*dmb2=P_[43],*dmWsc=P_[44],*dmbsc=P_[45];
  const float *plW1=P_[46],*plb1=P_[47],*plW2=P_[48],*plb2=P_[49],*plW3=P_[50],*plb3=P_[51];
  const float *mpos=P_[52];

  float* outPred   = (float*)d_out;                                   // (B,A,K,F,4)
  float* outScores = outPred + (int64_t)BATCH * AGT * KMODE * FHOR * 4;
  float* outPlan   = outScores + (int64_t)BATCH * AGT * KMODE;        // (B,F,3)

  // scratch carve (floats)
  float* wsf = (float*)d_ws; size_t off = 0;
  auto alloc = [&](size_t n) { float* p = wsf + off; off += (n + 63) & ~(size_t)63; return p; };
  float* bufK   = alloc((int64_t)BATCH * ENVL * DMODEL);
  float* bufV   = alloc((int64_t)BATCH * ENVL * DMODEL);
  float* bufQ   = alloc((int64_t)BATCH * AGT * DMODEL);
  float* bufCtx = alloc((int64_t)BATCH * AGT * DMODEL);
  float* bufLog = alloc((int64_t)BATCH * NHEAD * AGT * ENVL);
  float* bufAL  = alloc((int64_t)BATCH * AGT * DMODEL);
  float* bufAA  = alloc((int64_t)BATCH * AGT * DMODEL);
  float* bufCat = alloc((int64_t)BATCH * AGT * 2 * DMODEL);
  float* bufFus = alloc((int64_t)BATCH * AGT * 2 * DMODEL);
  float* bufInter = alloc((int64_t)BATCH * AGT * DMODEL);
  float* bufAtt = alloc((int64_t)BATCH * AGT * DMODEL);
  float* bufAtt2= alloc((int64_t)BATCH * AGT * DMODEL);
  float* bufSE  = alloc((int64_t)BATCH * AGT * DMODEL);
  float* bufTraj= alloc((int64_t)BATCH * AGT * KMODE * FHOR * 2);
  float* bufFutA= alloc((int64_t)BATCH * AGT * KMODE * DMODEL);
  float* bufFut = alloc((int64_t)BATCH * AGT * KMODE * DMODEL);
  float* bufSm  = alloc((int64_t)BATCH * AGT * KMODE);
  float* bufFutu= alloc((int64_t)BATCH * AGT * DMODEL);
  float* bufEnv = alloc((int64_t)BATCH * ENVL * DMODEL);
  unsigned char* envMask = (unsigned char*)alloc(BATCH * ENVL / 4 + 64);
  float* refMk  = alloc(BATCH * RREF);
  float* bufRefH= alloc((int64_t)BATCH * RREF * PPTS * DMODEL);
  float* bufXrM = alloc(BATCH * RREF * DMODEL);
  float* bufXr  = alloc(BATCH * RREF * DMODEL);
  float* bufXq  = alloc(BATCH * RREF * DMODEL);
  float* bufXr2 = alloc(BATCH * RREF * DMODEL);
  float* bufMlp = alloc(BATCH * RREF * DMODEL);
  float* scoreR = alloc(BATCH * RREF);
  float* bufEgo = alloc(BATCH * DMODEL);
  float* bufH1  = alloc(BATCH * 512);
  float* bufH2  = alloc(BATCH * DMODEL);
  float* bufCtl = alloc(BATCH * FHOR * 2);
  (void)ws_size; (void)in_sizes; (void)out_size;

  const int BA = BATCH * AGT;   // 672
  const int BR = BATCH * RREF;  // 192

  // ---- q = enc_agent @ ca.Wq (batched slice of encoding) ----
  { GemmParams p = GP();
    p.A = encoding; p.lda = DMODEL; p.sAb = (int64_t)LTOT * DMODEL;
    p.Bm = caWq; p.ldb = DMODEL;
    p.C = bufQ; p.ldc = DMODEL; p.sCb = (int64_t)AGT * DMODEL;
    p.M = AGT; p.N = DMODEL; p.K = DMODEL;
    launch_gemm(stream, p, BATCH); }
  // ---- al = MHA(enc_agent, enc_map, ca, map_mask) ----
  for (int t = 0; t < 2; t++) {
    GemmParams p = GP();
    p.A = encoding + (int64_t)AGT * DMODEL; p.lda = DMODEL; p.sAb = (int64_t)LTOT * DMODEL;
    p.Bm = t ? caWv : caWk; p.ldb = DMODEL;
    p.C = t ? bufV : bufK; p.ldc = DMODEL; p.sCb = (int64_t)MMAP * DMODEL;
    p.M = MMAP; p.N = DMODEL; p.K = DMODEL;
    launch_gemm(stream, p, BATCH);
  }
  attn_core(stream, bufQ, bufK, bufV, bufLog, bufCtx, AGT, MMAP, mapm);
  proj(stream, bufCtx, BA, caWo, bufAL);
  // ---- aa = MHA(enc_agent, enc_agent, ca, actors_mask) ----
  for (int t = 0; t < 2; t++) {
    GemmParams p = GP();
    p.A = encoding; p.lda = DMODEL; p.sAb = (int64_t)LTOT * DMODEL;
    p.Bm = t ? caWv : caWk; p.ldb = DMODEL;
    p.C = t ? bufV : bufK; p.ldc = DMODEL; p.sCb = (int64_t)AGT * DMODEL;
    p.M = AGT; p.N = DMODEL; p.K = DMODEL;
    launch_gemm(stream, p, BATCH);
  }
  attn_core(stream, bufQ, bufK, bufV, bufLog, bufCtx, AGT, AGT, actm);
  proj(stream, bufCtx, BA, caWo, bufAA);
  // ---- fusion MLP on concat(al, aa) ----
  k_concat2<<<grid1d((int64_t)BA * 512), 256, 0, stream>>>(bufAL, bufAA, bufCat, BA, DMODEL);
  proj(stream, bufCat, BA, fuW1, bufFus, fub1, nullptr, 1, 512, 512);
  proj(stream, bufFus, BA, fuW2, bufInter, fub2, nullptr, 0, DMODEL, 512);
  // ---- att = MHA(inter, al, mm, actors_mask) ----
  proj(stream, bufInter, BA, mmWq, bufQ);
  proj(stream, bufAL, BA, mmWk, bufK);
  proj(stream, bufAL, BA, mmWv, bufV);
  attn_core(stream, bufQ, bufK, bufV, bufLog, bufCtx, AGT, AGT, actm);
  proj(stream, bufCtx, BA, mmWo, bufAtt);
  // ---- 3x cross-attention to encoding (K/V hoisted: loop-invariant) ----
  proj(stream, encoding, BATCH * LTOT, itWk, bufK);
  proj(stream, encoding, BATCH * LTOT, itWv, bufV);
  float* attCur = bufAtt; float* attNxt = bufAtt2;
  for (int it = 0; it < 3; it++) {
    proj(stream, attCur, BA, itWq, bufQ);
    attn_core(stream, bufQ, bufK, bufV, bufLog, bufCtx, AGT, LTOT, msk);
    proj(stream, bufCtx, BA, itWo, attNxt, nullptr, attCur);  // residual fused
    float* t = attCur; attCur = attNxt; attNxt = t;
  }
  // ---- GMM head ----
  proj(stream, attCur, BA, gWt, outPred, gbt, nullptr, 0, KMODE * FHOR * 4, DMODEL);
  scalar_gemm<<<grid1d((int64_t)BA * KMODE), 256, 0, stream>>>(
      attCur, DMODEL, gWs, gbs, outScores, KMODE, BA, KMODE, DMODEL, 0);
  // ---- future encoder ----
  scalar_gemm<<<grid1d((int64_t)BA * DMODEL), 256, 0, stream>>>(
      actors + (TSTEP - 1) * SDIM, (int64_t)TSTEP * SDIM, feWx, febx,
      bufSE, DMODEL, BA, DMODEL, SDIM, 0);
  k_traj_pack<<<grid1d((int64_t)BA * KMODE * FHOR * 2), 256, 0, stream>>>(outPred, bufTraj);
  { GemmParams p = GP();                       // relu(traj@Wt + bt + state_emb[row/K])
    p.A = bufTraj; p.lda = FHOR * 2; p.Bm = feWt; p.ldb = DMODEL;
    p.C = bufFutA; p.ldc = DMODEL; p.bias = febt;
    p.add = bufSE; p.addld = DMODEL; p.addDiv = KMODE; p.act = 1;
    p.M = BA * KMODE; p.N = DMODEL; p.K = FHOR * 2;
    launch_gemm(stream, p, 1); }
  proj(stream, bufFutA, BA * KMODE, feWo, bufFut, febo);
  k_scores_softmax<<<grid1d(BA), 256, 0, stream>>>(outScores, bufSm, BA);
  k_futures<<<grid1d((int64_t)BA * DMODEL), 256, 0, stream>>>(bufFut, bufSm, bufFutu, BA, DMODEL);
  // ---- env = concat(futures, encoding) ----
  k_env_pack<<<grid1d((int64_t)BATCH * ENVL * DMODEL), 256, 0, stream>>>(bufFutu, encoding, bufEnv, DMODEL);
  k_mask_pack<<<grid1d(BATCH * ENVL), 256, 0, stream>>>(actm, msk, envMask);
  // ---- reference-path encoder ----
  scalar_gemm<<<grid1d((int64_t)BR * PPTS * DMODEL), 256, 0, stream>>>(
      ref_paths, 5, rW1, rb1, bufRefH, DMODEL, BR * PPTS, DMODEL, 5, 1);
  k_maxpool<<<grid1d((int64_t)BR * DMODEL), 256, 0, stream>>>(bufRefH, bufXrM, BR, PPTS, DMODEL);
  proj(stream, bufXrM, BR, rW2, bufXr, rb2);
  k_refmask<<<grid1d(BR), 256, 0, stream>>>(ref_paths, refMk);
  // ---- 4x decoder loop (env K/V hoisted: loop-invariant) ----
  proj(stream, bufEnv, BATCH * ENVL, dlWk, bufK);
  proj(stream, bufEnv, BATCH * ENVL, dlWv, bufV);
  for (int it = 0; it < 4; it++) {
    k_add_bcast<<<grid1d((int64_t)BR * DMODEL), 256, 0, stream>>>(bufXr, mpos, bufXq, BR, DMODEL);
    proj(stream, bufXq, BR, dlWq, bufQ);
    attn_core(stream, bufQ, bufK, bufV, bufLog, bufCtx, RREF, ENVL, envMask);
    proj(stream, bufCtx, BR, dlWo, bufXr2, nullptr, bufXr);          // xr + out
    proj(stream, bufXr2, BR, dmW1, bufMlp, dmb1, nullptr, 1);        // relu MLP
    proj(stream, bufMlp, BR, dmW2, bufXr, dmb2, bufXr2);             // + residual
  }
  k_score_r<<<grid1d(BR), 256, 0, stream>>>(bufXr, dmWsc, dmbsc, refMk, scoreR);
  k_argmax_gather<<<BATCH, 256, 0, stream>>>(scoreR, bufXr, bufEgo);
  // ---- planner MLP + dynamics rollout ----
  proj(stream, bufEgo, BATCH, plW1, bufH1, plb1, nullptr, 2, 512, DMODEL);
  proj(stream, bufH1, BATCH, plW2, bufH2, plb2, nullptr, 2, DMODEL, 512);
  proj(stream, bufH2, BATCH, plW3, bufCtl, plb3, nullptr, 0, FHOR * 2, DMODEL);
  k_dynamics<<<1, BATCH, 0, stream>>>(bufCtl, actors, outPlan);
}
